// BranchNet0_420906795043
// MI455X (gfx1250) — matrix operand order repaired and checked by reading, not compiled, not run on hardware
//
#include <hip/hip_runtime.h>

// ---------------------------------------------------------------------------
// BranchNet on MI455X (gfx1250): fused trunk + router, then routed classifier.
// - Heavy math: v_wmma_f32_16x16x32_bf16 (wave32 WMMA)
// - B matrices staged in LDS (padded rows, conflict-free ds_load_b128),
//   via the Tensor Data Mover (tensor_load_to_lds + s_wait_tensorcnt) when
//   the builtin is available, cooperative copy otherwise.
// ---------------------------------------------------------------------------

#define NROWS   131072
#define KDIM    256     // IN_DIM
#define HDIM    256     // HID
#define NBR     8       // B
#define CDIM    128     // C

#define LDS_STRIDE 264  // 256 bf16 + 8 pad -> 528 B rows (16B aligned, 64-bank spread)

typedef __attribute__((ext_vector_type(16))) __bf16 bf16x16;
typedef __attribute__((ext_vector_type(8)))  float  f32x8;
typedef __attribute__((ext_vector_type(4)))  unsigned int u32x4;
typedef __attribute__((ext_vector_type(4)))  int          i32x4;
typedef __attribute__((ext_vector_type(8)))  int          i32x8;

#if defined(__has_builtin)
#  if __has_builtin(__builtin_amdgcn_tensor_load_to_lds) && __has_builtin(__builtin_amdgcn_s_wait_tensorcnt)
#    define HAVE_TDM 1
#  endif
#endif
#ifndef HAVE_TDM
#  define HAVE_TDM 0
#endif

struct __align__(32) BfFrag {
    union {
        bf16x16        v;
        uint4          q[2];
        unsigned short u[16];
        unsigned int   d[8];
    };
};

__device__ __forceinline__ unsigned short f2bf(float f) {
    unsigned u = __float_as_uint(f);
    unsigned r = u + 0x7FFFu + ((u >> 16) & 1u);   // round-to-nearest-even
    return (unsigned short)(r >> 16);
}
__device__ __forceinline__ float bf2f(unsigned short s) {
    return __uint_as_float(((unsigned)s) << 16);
}

// packed fp32x2 -> bf16x2 (low = a, high = b)
#if defined(__has_builtin) && __has_builtin(__builtin_amdgcn_cvt_pk_bf16_f32)
typedef __attribute__((ext_vector_type(2))) __bf16 bf16x2;
__device__ __forceinline__ unsigned pk2bf(float a, float b) {
    bf16x2 t = __builtin_amdgcn_cvt_pk_bf16_f32(a, b);
    return __builtin_bit_cast(unsigned, t);
}
#else
__device__ __forceinline__ unsigned pk2bf(float a, float b) {
    return ((unsigned)f2bf(a)) | (((unsigned)f2bf(b)) << 16);
}
#endif

// B-fragment (32x16 bf16) from padded LDS tile: contiguous 32 B per lane.
// lanes 0-15: col = base col, K kk..kk+15 ; lanes 16-31: K kk+16..kk+31
__device__ __forceinline__ void load_bfrag_lds(BfFrag& d, const unsigned short* base,
                                               int col, int kb) {
    const uint4* p = (const uint4*)(base + col * LDS_STRIDE + kb);
    d.q[0] = p[0];
    d.q[1] = p[1];
}

// ---------------------------------------------------------------------------
// TDM stage of a [rows x 256] bf16 global tile into padded LDS rows
// (row stride 528 B): pad_enable, pad_interval=128 dwords, pad_amount=4 dwords.
// Issue from one wave, then s_wait_tensorcnt(0) + workgroup barrier.
// ---------------------------------------------------------------------------
#if HAVE_TDM
__device__ __forceinline__ void stage_tile_tdm(unsigned short* lds_dst,
                                               const unsigned short* gsrc, int rows) {
    const unsigned long long ga = (unsigned long long)(size_t)gsrc;
    u32x4 g0 = {
        1u,                                            // count=1 valid descriptor
        (unsigned)(size_t)lds_dst,                     // lds_addr (bytes, low 32 of flat)
        (unsigned)(ga & 0xFFFFFFFFull),                // global_addr[31:0]
        ((unsigned)((ga >> 32) & 0x01FFFFFFull)) | 0x80000000u // addr[56:32] | type=2
    };
    i32x8 g1 = {
        (int)((1u << 16) | (1u << 20) | (6u << 22) | (3u << 25)),
        // data_size=2B, pad_enable, pad_interval=128 DW (512 B), pad_amount=4 DW (16 B)
        (int)(256u << 16),        // tensor_dim0 = 256  (bits 79:48 -> low16 here)
        (int)(256u << 16),        // tensor_dim0 hi=0 ; tensor_dim1 = 256 low16
        (int)(256u << 16),        // tensor_dim1 hi=0 ; tile_dim0 = 256
        rows,                     // tile_dim1 = rows ; tile_dim2 = 0
        256,                      // tensor_dim0_stride = 256 (lo32)
        0,                        // stride hi ; tensor_dim1_stride lo16 = 0
        0
    };
    i32x4 g2 = {0, 0, 0, 0};
    i32x4 g3 = {0, 0, 0, 0};
#if defined(__clang_major__) && (__clang_major__ >= 23)
    i32x8 g45 = {0, 0, 0, 0, 0, 0, 0, 0};
    __builtin_amdgcn_tensor_load_to_lds(g0, g1, g2, g3, g45, 0);
#else
    __builtin_amdgcn_tensor_load_to_lds(g0, g1, g2, g3, 0);
#endif
}
#endif

// ---------------------------------------------------------------------------
// K0: weight transpose/convert + zero routing counters.
//   WbT[h][k]      = bf16(W_base[k][h])       (256x256)
//   WclfT[b][c][h] = bf16(W_clf[b][h][c])     (8x128x256)
// ---------------------------------------------------------------------------
__global__ __launch_bounds__(256) void prep_kernel(
    const float* __restrict__ W_base, const float* __restrict__ W_clf,
    unsigned short* __restrict__ WbT, unsigned short* __restrict__ WclfT,
    int* __restrict__ counts)
{
    int tid = blockIdx.x * 256 + threadIdx.x;
    if (tid < NBR) counts[tid] = 0;
    if (tid < HDIM * KDIM) {
        int h = tid >> 8, k = tid & 255;
        WbT[tid] = f2bf(W_base[k * HDIM + h]);
    }
    {   // 8*128*256 = 262144 elements, exactly one per thread
        int b   = tid >> 15;
        int rem = tid & 32767;
        int c   = rem >> 8, h = rem & 255;
        WclfT[tid] = f2bf(W_clf[((size_t)b * HDIM + h) * CDIM + c]);
    }
}

// ---------------------------------------------------------------------------
// K1: base_out GEMM (WMMA bf16, B from LDS) + bias + est/argmin + bf16 spill.
// 1024 blocks x 256 threads (8 waves); 128 rows per block, 16 per wave.
// LDS: WbT tile 135168 + bs 67584 + W_est 8192 + est 4096 = 215040 B
// ---------------------------------------------------------------------------
__global__ __launch_bounds__(256) void trunk_route_kernel(
    const float* __restrict__ x, const float* __restrict__ b_base,
    const float* __restrict__ W_est, const float* __restrict__ b_est,
    const unsigned short* __restrict__ WbT,
    unsigned short* __restrict__ baseBf,
    int* __restrict__ counts, int* __restrict__ rowlist)
{
    extern __shared__ char smem[];
    unsigned short* wbt_s  = (unsigned short*)smem;                          // 256x264 bf16
    unsigned short* bs     = (unsigned short*)(smem + 256 * LDS_STRIDE * 2); // 128x264 bf16
    float*          west_s = (float*)(smem + (256 + 128) * LDS_STRIDE * 2);  // 8x256 f32
    float*          est_s  = west_s + NBR * HDIM;                            // 8x128 f32

    const int tid  = threadIdx.x;
    const int lane = tid & 31;
    const int w    = tid >> 5;
    const int lrow = lane & 15;
    const bool hi  = lane >= 16;

    // ---- stage WbT (256x256 bf16) into padded LDS rows ----
#if HAVE_TDM
    if (w == 0) {
        stage_tile_tdm(wbt_s, WbT, 256);
        __builtin_amdgcn_s_wait_tensorcnt(0);
    }
#else
    for (int i = tid; i < 256 * 32; i += 256) {
        const int row = i >> 5, ch = i & 31;
        *(uint4*)(wbt_s + row * LDS_STRIDE + ch * 8) =
            *(const uint4*)(WbT + row * 256 + ch * 8);
    }
#endif
    for (int i = tid; i < NBR * HDIM; i += 256) west_s[i] = W_est[i];
    __syncthreads();

    const int rowBase = blockIdx.x * 128 + w * 16;
    const size_t xrow = (size_t)(rowBase + lrow) * KDIM;

    f32x8 acc[16];
#pragma unroll
    for (int i = 0; i < 16; ++i) {
#pragma unroll
        for (int j = 0; j < 8; ++j) acc[i][j] = 0.0f;
    }

    for (int kk = 0; kk < KDIM; kk += 32) {
        // A fragment: 16x32 bf16 from fp32 x (contiguous K: lanes 0-15 kk..kk+15, 16-31 kk+16..kk+31), packed converts
        BfFrag a;
        {
            const int k0 = kk + (hi ? 16 : 0);
            const float4* p0 = (const float4*)(x + xrow + k0);
            const float4* p1 = (const float4*)(x + xrow + k0 + 8);
            float4 f0 = p0[0], f1 = p0[1], f2 = p1[0], f3 = p1[1];
            a.d[0] = pk2bf(f0.x, f0.y);
            a.d[1] = pk2bf(f0.z, f0.w);
            a.d[2] = pk2bf(f1.x, f1.y);
            a.d[3] = pk2bf(f1.z, f1.w);
            a.d[4] = pk2bf(f2.x, f2.y);
            a.d[5] = pk2bf(f2.z, f2.w);
            a.d[6] = pk2bf(f3.x, f3.y);
            a.d[7] = pk2bf(f3.z, f3.w);
        }
        const int kb = kk + (hi ? 16 : 0);
#pragma unroll
        for (int nt = 0; nt < 16; ++nt) {
            BfFrag bf;
            load_bfrag_lds(bf, wbt_s, nt * 16 + lrow, kb);
            acc[nt] = __builtin_amdgcn_wmma_f32_16x16x32_bf16(
                false, a.v, false, bf.v, (short)0, acc[nt], false, false);
        }
    }

    // bias + store bf16 tile to LDS (C/D layout: comp r -> row r or r+8)
#pragma unroll
    for (int nt = 0; nt < 16; ++nt) {
        const int c  = nt * 16 + lrow;
        const float bb = b_base[c];
#pragma unroll
        for (int r = 0; r < 8; ++r) {
            const int m = w * 16 + r + (hi ? 8 : 0);
            bs[m * LDS_STRIDE + c] = f2bf(acc[nt][r] + bb);
        }
    }
    __syncthreads();

    // est partials: 2 threads per row, 4 branches each (all 256 threads busy)
    {
        const int row = tid & 127;
        const int bo  = (tid >> 7) * 4;              // 0 or 4
        const unsigned short* rowp = bs + row * LDS_STRIDE;
        float s0 = 0.f, s1 = 0.f, s2 = 0.f, s3 = 0.f;
        for (int h = 0; h < HDIM; ++h) {
            const float v = bf2f(rowp[h]);
            s0 += v * west_s[(bo + 0) * HDIM + h];
            s1 += v * west_s[(bo + 1) * HDIM + h];
            s2 += v * west_s[(bo + 2) * HDIM + h];
            s3 += v * west_s[(bo + 3) * HDIM + h];
        }
        est_s[(bo + 0) * 128 + row] = s0;
        est_s[(bo + 1) * 128 + row] = s1;
        est_s[(bo + 2) * 128 + row] = s2;
        est_s[(bo + 3) * 128 + row] = s3;
    }

    // coalesced LDS -> global bf16 spill of base_out (between the 2 barriers)
    for (int i = tid; i < 128 * 32; i += 256) {
        const int row = i >> 5, ch = i & 31;
        uint4 v = *(const uint4*)(bs + row * LDS_STRIDE + ch * 8);
        *(uint4*)(baseBf + ((size_t)(blockIdx.x * 128 + row)) * HDIM + ch * 8) = v;
    }
    __syncthreads();

    // argmin + routing scatter (one thread per row)
    if (tid < 128) {
        int best = 0;
        float bv = est_s[tid] + b_est[0];
#pragma unroll
        for (int b = 1; b < NBR; ++b) {
            const float sv = est_s[b * 128 + tid] + b_est[b];
            if (sv < bv) { bv = sv; best = b; }
        }
        const int grow = blockIdx.x * 128 + tid;
        const int pos  = atomicAdd(&counts[best], 1);
        rowlist[best * NROWS + pos] = grow;
    }
}

// ---------------------------------------------------------------------------
// K2: routed classifier. grid = 8 branches x 1024 chunks; 8 waves/block,
// 16 gathered rows per wave, C=128 -> 8 WMMA N-tiles, fp32 scatter out.
// Branch weights (128x256 bf16) staged in LDS via TDM.
// ---------------------------------------------------------------------------
__global__ __launch_bounds__(256) void classifier_kernel(
    const unsigned short* __restrict__ baseBf,
    const unsigned short* __restrict__ WclfT,
    const float* __restrict__ b_clf,
    const int* __restrict__ counts, const int* __restrict__ rowlist,
    float* __restrict__ out)
{
    extern __shared__ char smem2[];
    unsigned short* wclf_s = (unsigned short*)smem2;          // 128x264 bf16 = 67584 B

    const int b     = blockIdx.x >> 10;
    const int chunk = blockIdx.x & 1023;
    const int cnt   = counts[b];
    if (chunk * 128 >= cnt) return;           // block-uniform early exit

    const int tid  = threadIdx.x;
    const int lane = tid & 31;
    const int w    = tid >> 5;
    const int lrow = lane & 15;
    const bool hi  = lane >= 16;

    const unsigned short* wb = WclfT + (size_t)b * CDIM * HDIM;

    // ---- stage branch weights (128x256 bf16) into padded LDS rows ----
#if HAVE_TDM
    if (w == 0) {
        stage_tile_tdm(wclf_s, wb, 128);
        __builtin_amdgcn_s_wait_tensorcnt(0);
    }
#else
    for (int i = tid; i < 128 * 32; i += 256) {
        const int row = i >> 5, ch = i & 31;
        *(uint4*)(wclf_s + row * LDS_STRIDE + ch * 8) =
            *(const uint4*)(wb + row * 256 + ch * 8);
    }
#endif
    __syncthreads();

    const int slot0 = chunk * 128 + w * 16;
    if (slot0 >= cnt) return;                 // uniform per wave: EXEC stays full
    int nrows = cnt - slot0;
    if (nrows > 16) nrows = 16;

    const int* lst = rowlist + b * NROWS;
    const int slot = slot0 + (lrow < nrows ? lrow : nrows - 1);
    const size_t arow = (size_t)lst[slot] * HDIM;

    f32x8 acc[8];
#pragma unroll
    for (int i = 0; i < 8; ++i) {
#pragma unroll
        for (int j = 0; j < 8; ++j) acc[i][j] = 0.0f;
    }

    for (int kk = 0; kk < HDIM; kk += 32) {
        const int kb = kk + (hi ? 16 : 0);
        BfFrag a;
        {
            const int k0 = kk + (hi ? 16 : 0);
            a.q[0] = *(const uint4*)(baseBf + arow + k0);
            a.q[1] = *(const uint4*)(baseBf + arow + k0 + 8);
        }
#pragma unroll
        for (int nt = 0; nt < 8; ++nt) {
            BfFrag bf;
            load_bfrag_lds(bf, wclf_s, nt * 16 + lrow, kb);
            acc[nt] = __builtin_amdgcn_wmma_f32_16x16x32_bf16(
                false, a.v, false, bf.v, (short)0, acc[nt], false, false);
        }
    }

    // bias + masked scatter to d_out (row per C/D component)
#pragma unroll
    for (int nt = 0; nt < 8; ++nt) {
        const int c = nt * 16 + lrow;
        const float bias = b_clf[b * CDIM + c];
#pragma unroll
        for (int r = 0; r < 8; ++r) {
            const int m = r + (hi ? 8 : 0);
            if (m < nrows) {
                const int orow = lst[slot0 + m];
                out[(size_t)orow * CDIM + c] = acc[nt][r] + bias;
            }
        }
    }
}

// ---------------------------------------------------------------------------
// Host-side launcher
// ---------------------------------------------------------------------------
extern "C" void kernel_launch(void* const* d_in, const int* in_sizes, int n_in,
                              void* d_out, int out_size, void* d_ws, size_t ws_size,
                              hipStream_t stream) {
    const float* x      = (const float*)d_in[0];
    const float* W_base = (const float*)d_in[1];
    const float* b_base = (const float*)d_in[2];
    const float* W_clf  = (const float*)d_in[3];
    const float* b_clf  = (const float*)d_in[4];
    const float* W_est  = (const float*)d_in[5];
    const float* b_est  = (const float*)d_in[6];
    float* out = (float*)d_out;

    // workspace layout
    char* ws = (char*)d_ws;
    int* counts            = (int*)ws;                                   //    32 B
    int* rowlist           = (int*)(ws + 256);                           //   4 MB
    unsigned short* WbT    = (unsigned short*)(ws + 256 + (size_t)NBR * NROWS * 4);
    unsigned short* WclfT  = WbT + (size_t)HDIM * KDIM;                  // 128 KB after WbT
    unsigned short* baseBf = WclfT + (size_t)NBR * CDIM * HDIM;          // 64 MB bf16 base_out

    prep_kernel<<<1024, 256, 0, stream>>>(W_base, W_clf, WbT, WclfT, counts);

    const size_t smem1 = (size_t)(256 + 128) * LDS_STRIDE * 2            // WbT tile + bs tile
                       + (size_t)NBR * HDIM * 4                          // W_est
                       + (size_t)NBR * 128 * 4;                          // est partials
    trunk_route_kernel<<<NROWS / 128, 256, smem1, stream>>>(
        x, b_base, W_est, b_est, WbT, baseBf, counts, rowlist);

    const size_t smem2 = (size_t)128 * LDS_STRIDE * 2;                   // branch weight tile
    classifier_kernel<<<NBR * 1024, 256, smem2, stream>>>(
        baseBf, WclfT, b_clf, counts, rowlist, out);
}
